// LocalStridedBlockSparseAttn_48825188221191
// MI455X (gfx1250) — matrix operand order repaired and checked by reading, not compiled, not run on hardware
//
#include <hip/hip_runtime.h>
#include <hip/hip_fp16.h>

// CDNA5 / gfx1250 wave32 WMMA types
typedef __attribute__((ext_vector_type(16))) _Float16 v16h;
typedef __attribute__((ext_vector_type(8)))  float    v8f;
typedef __attribute__((ext_vector_type(4)))  float    f32x4;

#define N_HEADS 16
#define HEAD    64
#define SEQ     4096
#define BS      64
#define LOCAL   8
#define VERT    8
#define NB      (SEQ / BS)
#define SM_SCALE 0.125f   // 1/sqrt(64)

union H16 {
  v16h     v;
  f32x4    f4[2];   // two 16-byte chunks (8 halfs each)
  _Float16 h[16];
};
union H8 {
  f32x4    f4;      // one 16-byte chunk
  _Float16 h[8];
};

// hard scheduler fence: nothing (incl. WMMA / VALU) may be moved across.
// This is what actually pins ds_load batches ahead of the WMMA chains —
// a plain asm memory clobber only orders memory ops, and WMMA is not one.
#define SCHED_BARRIER() __builtin_amdgcn_sched_barrier(0)

__device__ __forceinline__ bool admit_blk(int qb, int c, int h) {
  return ((qb - c) < LOCAL) || (((c + h + 1) % VERT) == 0);
}

__global__ __launch_bounds__(128)
void bsattn_f16wmma_kernel(const float* __restrict__ q,
                           const float* __restrict__ k,
                           const float* __restrict__ v,
                           float* __restrict__ out)
{
  const int h   = blockIdx.x & (N_HEADS - 1);   // head
  const int qb  = blockIdx.x >> 4;              // query block 0..63
  const int tid = threadIdx.x;
  const int lane  = tid & 31;
  const int wv    = tid >> 5;      // wave id = M-tile (16 rows each)
  const int lhalf = lane >> 4;     // 0 / 1 (lane half)
  const int l16   = lane & 15;

  // LDS: K block (row j, col d), V^T block (row d, col j), P per wave (row M, col j)
  __shared__ alignas(16) _Float16 Ks[BS][HEAD];   // 8 KB
  __shared__ alignas(16) _Float16 Vt[HEAD][BS];   // 8 KB
  __shared__ alignas(16) _Float16 Ps[4][16][BS];  // 8 KB

  // ---------------- Q A-tiles (persistent across k-blocks) -----------------
  // A 16x32 f16 layout: lane<16 : M=lane,    K = {0..7, 16..23}
  //                     lane>=16: M=lane-16, K = {8..15, 24..31}
  v16h aQ[2];
  {
    const int row = qb * BS + wv * 16 + l16;
    const float* qrow = q + ((size_t)row * N_HEADS + h) * HEAD;
    #pragma unroll
    for (int t = 0; t < 2; ++t) {
      H16 a;
      #pragma unroll
      for (int g = 0; g < 2; ++g) {
        const int d0 = 32 * t + g * 16 + lhalf * 8;
        #pragma unroll
        for (int e = 0; e < 8; ++e)
          a.h[g * 8 + e] = (_Float16)qrow[d0 + e];
      }
      aQ[t] = a.v;
    }
  }

  // online softmax state: each lane holds 8 rows (C/D layout: M = r + 8*lhalf)
  float m_run[8], l_run[8];
  v8f acc[4];
  const v8f vzero = {0.f, 0.f, 0.f, 0.f, 0.f, 0.f, 0.f, 0.f};
  #pragma unroll
  for (int r = 0; r < 8; ++r) { m_run[r] = -3.0e38f; l_run[r] = 0.f; }
  #pragma unroll
  for (int n = 0; n < 4; ++n) acc[n] = vzero;

  // staging work partition (constant per thread)
  const int krow_i = tid >> 1;          // K: one row per 2 threads
  const int kcb    = (tid & 1) * 32;    // 32 contiguous floats
  const int vjp    = (tid >> 2) * 2;    // V: pair of rows j, j+1
  const int vcb    = (tid & 3) * 16;    // 16 contiguous floats each

  // ---------------- iterate admitted k-blocks (uniform per WG) -------------
  for (int c = 0; c <= qb; ++c) {
    if (!admit_blk(qb, c, h)) continue;

    __syncthreads();  // previous iteration's LDS readers done
    {
      // ---- stage K (row-major f16, packed b128 stores) ----
      const float* krow = k + ((size_t)(c * BS + krow_i) * N_HEADS + h) * HEAD + kcb;
      #pragma unroll
      for (int ch = 0; ch < 4; ++ch) {
        H8 pk;
        #pragma unroll
        for (int e = 0; e < 8; ++e) pk.h[e] = (_Float16)krow[ch * 8 + e];
        *(f32x4*)&Ks[krow_i][kcb + ch * 8] = pk.f4;
      }
      // ---- stage V^T (pairs of j-rows -> packed half2 b32 stores) ----
      const float* vr0 = v + ((size_t)(c * BS + vjp)     * N_HEADS + h) * HEAD + vcb;
      const float* vr1 = v + ((size_t)(c * BS + vjp + 1) * N_HEADS + h) * HEAD + vcb;
      #pragma unroll
      for (int e = 0; e < 16; ++e) {
        const __half2 p2 = __floats2half2_rn(vr0[e], vr1[e]);
        *(__half2*)&Vt[vcb + e][vjp] = p2;
      }

      // ---- prefetch next admitted block's K/V (global_prefetch_b8) ----
      int cn = c + 1;
      while (cn <= qb && !admit_blk(qb, cn, h)) ++cn;
      if (cn <= qb) {
        __builtin_prefetch(k + ((size_t)(cn * BS + krow_i) * N_HEADS + h) * HEAD + kcb, 0, 1);
        __builtin_prefetch(v + ((size_t)(cn * BS + krow_i) * N_HEADS + h) * HEAD + kcb, 0, 1);
      }
    }
    __syncthreads();

    // ---- S = Q * K^T : issue ALL 16 ds_load_b128, hard barrier, then WMMAs ----
    // B 32x16 f16: lane<16: N=lane, K={0..7,16..23}; lane>=16: N=lane-16, K={8..15,24..31}
    v8f s[4];
    {
      H16 b0[4], b1[4];
      #pragma unroll
      for (int n2 = 0; n2 < 4; ++n2) {
        const _Float16* src = &Ks[n2 * 16 + l16][lhalf * 8];
        b0[n2].f4[0] = *(const f32x4*)(src);
        b0[n2].f4[1] = *(const f32x4*)(src + 16);
        b1[n2].f4[0] = *(const f32x4*)(src + 32);
        b1[n2].f4[1] = *(const f32x4*)(src + 48);
      }
      SCHED_BARRIER();   // loads stay batched; WMMAs wait on decreasing dscnt
      #pragma unroll
      for (int n2 = 0; n2 < 4; ++n2)
        s[n2] = __builtin_amdgcn_wmma_f32_16x16x32_f16(
                    false, aQ[0], false, b0[n2].v, (short)0, vzero, false, false);
      #pragma unroll
      for (int n2 = 0; n2 < 4; ++n2)
        s[n2] = __builtin_amdgcn_wmma_f32_16x16x32_f16(
                    false, aQ[1], false, b1[n2].v, (short)0, s[n2], false, false);
    }
    SCHED_BARRIER();

    // ---- hoist V^T B-tile loads: they drain from LDS under the softmax ----
    H16 bv0[4], bv1[4];
    #pragma unroll
    for (int n2 = 0; n2 < 4; ++n2) {
      const _Float16* src = &Vt[n2 * 16 + l16][lhalf * 8];
      bv0[n2].f4[0] = *(const f32x4*)(src);
      bv0[n2].f4[1] = *(const f32x4*)(src + 16);
      bv1[n2].f4[0] = *(const f32x4*)(src + 32);
      bv1[n2].f4[1] = *(const f32x4*)(src + 48);
    }
    SCHED_BARRIER();   // V loads issued before the softmax VALU block

    // ---- scale + causal mask on diagonal block ----
    #pragma unroll
    for (int n2 = 0; n2 < 4; ++n2) {
      #pragma unroll
      for (int r = 0; r < 8; ++r) {
        float sv = s[n2][r] * SM_SCALE;
        if (c == qb) {
          const int M = wv * 16 + r + 8 * lhalf;  // q row within block
          const int j = n2 * 16 + l16;            // k pos within block
          if (j > M) sv = -3.0e38f;
        }
        s[n2][r] = sv;
      }
    }

    // ---- online softmax (rows live across the 16 lanes of a half) ----
    #pragma unroll
    for (int r = 0; r < 8; ++r) {
      float mx = fmaxf(fmaxf(s[0][r], s[1][r]), fmaxf(s[2][r], s[3][r]));
      #pragma unroll
      for (int off = 1; off < 16; off <<= 1)
        mx = fmaxf(mx, __shfl_xor(mx, off, 32));

      const float m_new = fmaxf(m_run[r], mx);
      const float corr  = __expf(m_run[r] - m_new);
      m_run[r] = m_new;
      l_run[r] *= corr;
      #pragma unroll
      for (int n2 = 0; n2 < 4; ++n2) acc[n2][r] *= corr;

      float rs = 0.f;
      const int M = r + 8 * lhalf;
      #pragma unroll
      for (int n2 = 0; n2 < 4; ++n2) {
        const float p = __expf(s[n2][r] - m_new);
        rs += p;
        Ps[wv][M][n2 * 16 + l16] = (_Float16)p;   // same-wave LDS, in-order
      }
      #pragma unroll
      for (int off = 1; off < 16; off <<= 1)
        rs += __shfl_xor(rs, off, 32);
      l_run[r] += rs;
    }

    // ---- O += P * V : read P back in A layout, V tiles already resident ----
    v16h aP[2];
    #pragma unroll
    for (int t = 0; t < 2; ++t) {
      H16 a;
      const _Float16* base = &Ps[wv][l16][0];
      a.f4[0] = *(const f32x4*)(base + 32 * t + lhalf * 8);
      a.f4[1] = *(const f32x4*)(base + 32 * t + 16 + lhalf * 8);
      aP[t] = a.v;
    }
    SCHED_BARRIER();
    #pragma unroll
    for (int n2 = 0; n2 < 4; ++n2)
      acc[n2] = __builtin_amdgcn_wmma_f32_16x16x32_f16(
                    false, aP[0], false, bv0[n2].v, (short)0, acc[n2], false, false);
    #pragma unroll
    for (int n2 = 0; n2 < 4; ++n2)
      acc[n2] = __builtin_amdgcn_wmma_f32_16x16x32_f16(
                    false, aP[1], false, bv1[n2].v, (short)0, acc[n2], false, false);
  }

  // ---------------- normalize + store fp32 output --------------------------
  #pragma unroll
  for (int r = 0; r < 8; ++r) {
    const float inv = 1.0f / l_run[r];
    const int row = qb * BS + wv * 16 + r + 8 * lhalf;
    float* orow = out + ((size_t)row * N_HEADS + h) * HEAD;
    #pragma unroll
    for (int n2 = 0; n2 < 4; ++n2)
      orow[n2 * 16 + l16] = acc[n2][r] * inv;
  }
}

extern "C" void kernel_launch(void* const* d_in, const int* in_sizes, int n_in,
                              void* d_out, int out_size, void* d_ws, size_t ws_size,
                              hipStream_t stream) {
  (void)in_sizes; (void)n_in; (void)out_size; (void)d_ws; (void)ws_size;
  const float* q = (const float*)d_in[0];
  const float* k = (const float*)d_in[1];
  const float* v = (const float*)d_in[2];
  // d_in[3] = cu_seqlens_k ([0, SEQ]) — single prefill sequence, unused.
  float* out = (float*)d_out;

  bsattn_f16wmma_kernel<<<dim3(N_HEADS * NB), dim3(128), 0, stream>>>(q, k, v, out);
}
